// GCN_19645180412674
// MI455X (gfx1250) — matrix operand order repaired and checked by reading, not compiled, not run on hardware
//
#include <hip/hip_runtime.h>

// ---------------------------------------------------------------------------
// Types for CDNA5 WMMA (wave32): v_wmma_f32_16x16x32_bf16
// ---------------------------------------------------------------------------
typedef __attribute__((ext_vector_type(16))) __bf16 v16bf;
typedef __attribute__((ext_vector_type(8)))  float  v8f;

union BF16x16 {
  v16bf v;
  unsigned short u[16];
  uint4 q[2];
};

__device__ __forceinline__ unsigned short f2bf(float f) {
  unsigned int u = __float_as_uint(f);
  u += 0x7FFFu + ((u >> 16) & 1u);   // round-to-nearest-even
  return (unsigned short)(u >> 16);
}

// ---------------------------------------------------------------------------
// Utility kernels
// ---------------------------------------------------------------------------
__global__ void gcn_zero_f32(float* __restrict__ p, long long n) {
  long long i = (long long)blockIdx.x * blockDim.x + threadIdx.x;
  if (i < n) p[i] = 0.0f;
}

__global__ void gcn_degree(const int* __restrict__ dst, float* __restrict__ deg,
                           long long E) {
  long long e = (long long)blockIdx.x * blockDim.x + threadIdx.x;
  if (e < E) atomicAdd(&deg[dst[e]], 1.0f);
}

__global__ void gcn_norm(const float* __restrict__ deg, float* __restrict__ dis,
                         float* __restrict__ selfn, int N) {
  int i = blockIdx.x * blockDim.x + threadIdx.x;
  if (i < N) {
    float d = deg[i] + 1.0f;          // +1 for self loop
    dis[i]   = rsqrtf(d);
    selfn[i] = 1.0f / d;
  }
}

// per-edge symmetric norm, streamed once and reused by both layers
__global__ void gcn_enorm(const int* __restrict__ src, const int* __restrict__ dst,
                          const float* __restrict__ dis, float* __restrict__ enorm,
                          long long E) {
  long long e = (long long)blockIdx.x * blockDim.x + threadIdx.x;
  if (e < E) enorm[e] = dis[src[e]] * dis[dst[e]];
}

__global__ void gcn_cvt_x(const float* __restrict__ x, unsigned short* __restrict__ xb,
                          long long n) {
  long long i = (long long)blockIdx.x * blockDim.x + threadIdx.x;
  if (i < n) xb[i] = f2bf(x[i]);
}

// W1 [128][64] f32 row-major  ->  W1t bf16, column-major [64 cols][128 k]
__global__ void gcn_cvt_w1(const float* __restrict__ W1, unsigned short* __restrict__ W1t) {
  int i = blockIdx.x * blockDim.x + threadIdx.x;     // 128*64
  if (i < 128 * 64) {
    int k = i >> 6, c = i & 63;
    W1t[c * 128 + k] = f2bf(W1[i]);
  }
}

// W2 [64][2] f32 -> W2t bf16 padded to 16 columns, column-major [16][64]
__global__ void gcn_cvt_w2(const float* __restrict__ W2, unsigned short* __restrict__ W2t) {
  int i = blockIdx.x * blockDim.x + threadIdx.x;     // 16*64
  if (i < 16 * 64) {
    int c = i >> 6, k = i & 63;
    W2t[i] = (c < 2) ? f2bf(W2[k * 2 + c]) : (unsigned short)0;
  }
}

// ---------------------------------------------------------------------------
// GEMM1: Xb[N][128] bf16 @ W1t(col-major [64][128]) -> H1[N][64] f32
// W1t staged to LDS once per block, hoisted to registers per wave, then each
// wave grid-strides over 16-row tiles: 8 A-loads + 16 WMMAs per tile.
// Store guard is hoisted to a uniform per-tile branch (only the last ragged
// tile takes the predicated path).
// ---------------------------------------------------------------------------
__global__ __launch_bounds__(256)
void gcn_gemm1(const unsigned short* __restrict__ Xb,
               const unsigned short* __restrict__ W1t,
               float* __restrict__ H1, int N, int ntiles) {
  __shared__ unsigned short sW[64 * 128];              // 16 KB of 320 KB/WGP

  // cooperative stage: 1024 x 16B chunks across 256 threads
  {
    const uint4* g = (const uint4*)W1t;
    uint4*       s = (uint4*)sW;
#pragma unroll
    for (int i = 0; i < 4; ++i)
      s[threadIdx.x + i * 256] = g[threadIdx.x + i * 256];
  }
  __syncthreads();

  const int lane = threadIdx.x & 31;
  const int wave = threadIdx.x >> 5;
  const int r    = lane & 15;
  const int half = lane >> 4;

  // Hoist entire B operand (16 KB) into registers: 16 x BF16x16 per lane.
  // B k order used here (contiguous, same as A): lanes 0-15 hold K[0..15], lanes 16-31 hold K[16..31].
  BF16x16 b[4][4];
#pragma unroll
  for (int kc = 0; kc < 4; ++kc)
#pragma unroll
    for (int nt = 0; nt < 4; ++nt) {
      const unsigned short* p = sW + (nt * 16 + r) * 128 + kc * 32 + half * 16;
      b[kc][nt].q[0] = *(const uint4*)(p);             // ds_load_b128
      b[kc][nt].q[1] = *(const uint4*)(p + 8);
    }

  const int step = gridDim.x * 8;
  for (int tile = blockIdx.x * 8 + wave; tile < ntiles; tile += step) {
    const int m0 = tile * 16;
    int row = m0 + r;
    if (row > N - 1) row = N - 1;                      // clamp: keep EXEC all-ones
    const unsigned short* xrow = Xb + (size_t)row * 128;

    v8f acc0 = {}, acc1 = {}, acc2 = {}, acc3 = {};
#pragma unroll
    for (int kc = 0; kc < 4; ++kc) {
      BF16x16 a;
      // A k order used here (contiguous, same as B): lanes 0-15 -> K[0..15]; 16-31 -> K[16..31]
      a.q[0] = *(const uint4*)(xrow + kc * 32 + half * 16);
      a.q[1] = *(const uint4*)(xrow + kc * 32 + half * 16 + 8);
      acc0 = __builtin_amdgcn_wmma_f32_16x16x32_bf16(false, a.v, false, b[kc][0].v,
                                                     (short)0, acc0, false, false);
      acc1 = __builtin_amdgcn_wmma_f32_16x16x32_bf16(false, a.v, false, b[kc][1].v,
                                                     (short)0, acc1, false, false);
      acc2 = __builtin_amdgcn_wmma_f32_16x16x32_bf16(false, a.v, false, b[kc][2].v,
                                                     (short)0, acc2, false, false);
      acc3 = __builtin_amdgcn_wmma_f32_16x16x32_bf16(false, a.v, false, b[kc][3].v,
                                                     (short)0, acc3, false, false);
    }

    // C/D layout: lanes 0-15 -> M = m0+0..7, lanes 16-31 -> M = m0+8..15, N = r
    v8f accs[4] = {acc0, acc1, acc2, acc3};
    float* hp = H1 + (size_t)(m0 + half * 8) * 64 + r;
    if (m0 + 16 <= N) {                                // full tile: uniform fast path
#pragma unroll
      for (int nt = 0; nt < 4; ++nt)
#pragma unroll
        for (int rr = 0; rr < 8; ++rr)
          hp[(size_t)rr * 64 + nt * 16] = accs[nt][rr];
    } else {                                           // single ragged tile
#pragma unroll
      for (int nt = 0; nt < 4; ++nt)
#pragma unroll
        for (int rr = 0; rr < 8; ++rr)
          if (m0 + half * 8 + rr < N)
            hp[(size_t)rr * 64 + nt * 16] = accs[nt][rr];
    }
  }
}

// ---------------------------------------------------------------------------
// GEMM2: Hb[N][64] bf16 @ W2t(col-major [16][64], cols>=2 zero) -> H2[N][2]
// B (2 KB) register-hoisted once per wave; grid-stride over tiles.
// ---------------------------------------------------------------------------
__global__ __launch_bounds__(256)
void gcn_gemm2(const unsigned short* __restrict__ Hb,
               const unsigned short* __restrict__ W2t,
               float* __restrict__ H2, int N, int ntiles) {
  const int lane = threadIdx.x & 31;
  const int wave = threadIdx.x >> 5;
  const int r    = lane & 15;
  const int half = lane >> 4;

  BF16x16 b[2];
#pragma unroll
  for (int kc = 0; kc < 2; ++kc) {
    const unsigned short* p = W2t + (size_t)r * 64 + kc * 32 + half * 16;
    b[kc].q[0] = *(const uint4*)(p);
    b[kc].q[1] = *(const uint4*)(p + 8);
  }

  const int step = gridDim.x * 8;
  for (int tile = blockIdx.x * 8 + wave; tile < ntiles; tile += step) {
    const int m0 = tile * 16;
    int row = m0 + r;
    if (row > N - 1) row = N - 1;
    const unsigned short* hrow = Hb + (size_t)row * 64;

    v8f acc = {};
#pragma unroll
    for (int kc = 0; kc < 2; ++kc) {
      BF16x16 a;
      a.q[0] = *(const uint4*)(hrow + kc * 32 + half * 16);
      a.q[1] = *(const uint4*)(hrow + kc * 32 + half * 16 + 8);
      acc = __builtin_amdgcn_wmma_f32_16x16x32_bf16(false, a.v, false, b[kc].v,
                                                    (short)0, acc, false, false);
    }

    float* hp = H2 + (size_t)(m0 + half * 8) * 2 + r;
    if (m0 + 16 <= N) {                                // full tile
      if (r < 2) {                                     // only real output columns
#pragma unroll
        for (int rr = 0; rr < 8; ++rr)
          hp[(size_t)rr * 2] = acc[rr];
      }
    } else {
      if (r < 2) {
#pragma unroll
        for (int rr = 0; rr < 8; ++rr)
          if (m0 + half * 8 + rr < N)
            hp[(size_t)rr * 2] = acc[rr];
      }
    }
  }
}

// ---------------------------------------------------------------------------
// Edge aggregation, layer 1: agg[dst][c..c+3] += H1[src][c..c+3] * enorm[e]
// One thread per (edge, 4-channel group): float4 gather + 4 L2 atomics.
// ---------------------------------------------------------------------------
__global__ void gcn_agg1(const int* __restrict__ src, const int* __restrict__ dst,
                         const float* __restrict__ enorm,
                         const float* __restrict__ H1, float* __restrict__ agg,
                         long long E) {
  long long t = (long long)blockIdx.x * blockDim.x + threadIdx.x;
  if (t >= E * 16) return;
  int e = (int)(t >> 4);
  int c = (int)(t & 15) * 4;
  int s = src[e], d = dst[e];
  if ((t & 15) == 0 && e + 4096 < E)                   // lookahead: global_prefetch_b8
    __builtin_prefetch(&H1[(size_t)src[e + 4096] * 64], 0, 0);
  float w = enorm[e];
  const float4 v = *(const float4*)&H1[(size_t)s * 64 + c];
  float* dp = &agg[(size_t)d * 64 + c];
  atomicAdd(dp + 0, v.x * w);
  atomicAdd(dp + 1, v.y * w);
  atomicAdd(dp + 2, v.z * w);
  atomicAdd(dp + 3, v.w * w);
}

// self-loop + bias + ReLU, emit bf16 activations for layer-2 WMMA
__global__ void gcn_post1(const float* __restrict__ agg, const float* __restrict__ H1,
                          const float* __restrict__ selfn, const float* __restrict__ b1,
                          unsigned short* __restrict__ Hb, int N) {
  long long t = (long long)blockIdx.x * blockDim.x + threadIdx.x;
  if (t >= (long long)N * 64) return;
  int i = (int)(t >> 6);
  int c = (int)(t & 63);
  float v = agg[t] + H1[t] * selfn[i] + b1[c];
  Hb[t] = f2bf(fmaxf(v, 0.0f));
}

// Edge aggregation, layer 2 (2 channels per edge thread)
__global__ void gcn_agg2(const int* __restrict__ src, const int* __restrict__ dst,
                         const float* __restrict__ enorm,
                         const float* __restrict__ H2, float* __restrict__ agg,
                         long long E) {
  long long e = (long long)blockIdx.x * blockDim.x + threadIdx.x;
  if (e >= E) return;
  int s = src[e], d = dst[e];
  float w = enorm[e];
  const float2 v = *(const float2*)&H2[(size_t)s * 2];
  float* dp = &agg[(size_t)d * 2];
  atomicAdd(dp + 0, v.x * w);
  atomicAdd(dp + 1, v.y * w);
}

__global__ void gcn_post2(const float* __restrict__ agg, const float* __restrict__ H2,
                          const float* __restrict__ selfn, const float* __restrict__ b2,
                          float* __restrict__ out, int N) {
  long long t = (long long)blockIdx.x * blockDim.x + threadIdx.x;
  if (t >= (long long)N * 2) return;
  int i = (int)(t >> 1);
  int c = (int)(t & 1);
  out[t] = agg[t] + H2[t] * selfn[i] + b2[c];
}

// ---------------------------------------------------------------------------
// Orchestration
// ---------------------------------------------------------------------------
static inline unsigned int nblk(long long n, int b) {
  return (unsigned int)((n + b - 1) / b);
}

extern "C" void kernel_launch(void* const* d_in, const int* in_sizes, int n_in,
                              void* d_out, int out_size, void* d_ws, size_t ws_size,
                              hipStream_t stream) {
  const float* x     = (const float*)d_in[0];
  const int*   edges = (const int*)d_in[1];
  const float* W1    = (const float*)d_in[2];
  const float* b1    = (const float*)d_in[3];
  const float* W2    = (const float*)d_in[4];
  const float* b2    = (const float*)d_in[5];
  float*       out   = (float*)d_out;

  const int       N = in_sizes[0] / 128;       // 100000
  const long long E = in_sizes[1] / 2;         // 3200000
  const int* src = edges;
  const int* dst = edges + E;

  // ---- carve workspace (256B aligned) ----
  char*  base = (char*)d_ws;
  size_t off  = 0;
  auto carve = [&](size_t bytes) -> char* {
    char* p = base + off;
    off = (off + bytes + 255) & ~(size_t)255;
    return p;
  };
  float*          deg   = (float*)carve((size_t)N * 4);
  float*          dis   = (float*)carve((size_t)N * 4);
  float*          selfn = (float*)carve((size_t)N * 4);
  float*          enorm = (float*)carve((size_t)E * 4);
  unsigned short* W1t   = (unsigned short*)carve(64 * 128 * 2);
  unsigned short* W2t   = (unsigned short*)carve(16 * 64 * 2);
  float*          H1    = (float*)carve((size_t)N * 64 * 4);
  unsigned short* Hb    = (unsigned short*)carve((size_t)N * 64 * 2);
  float*          H2    = (float*)carve((size_t)N * 2 * 4);
  float*          agg2  = (float*)carve((size_t)N * 2 * 4);
  // Xb (bf16 N*128) and agg1 (f32 N*64) have disjoint live ranges: alias them.
  char*           shared = carve((size_t)N * 256);
  unsigned short* Xb   = (unsigned short*)shared;
  float*          agg1 = (float*)shared;

  const int B = 256;
  const int ntiles = (N + 15) / 16;
  unsigned int gemmGrid = nblk(ntiles, 8);
  if (gemmGrid > 256) gemmGrid = 256;          // grid-stride: ~3 tiles per wave

  // degrees + norms
  gcn_zero_f32<<<nblk(N, B), B, 0, stream>>>(deg, N);
  gcn_degree <<<nblk(E, B), B, 0, stream>>>(dst, deg, E);
  gcn_norm   <<<nblk(N, B), B, 0, stream>>>(deg, dis, selfn, N);
  gcn_enorm  <<<nblk(E, B), B, 0, stream>>>(src, dst, dis, enorm, E);

  // bf16 conversions
  gcn_cvt_x <<<nblk((long long)N * 128, B), B, 0, stream>>>(x, Xb, (long long)N * 128);
  gcn_cvt_w1<<<nblk(128 * 64, B), B, 0, stream>>>(W1, W1t);
  gcn_cvt_w2<<<nblk(16 * 64, B), B, 0, stream>>>(W2, W2t);

  // layer 1
  gcn_gemm1<<<gemmGrid, B, 0, stream>>>(Xb, W1t, H1, N, ntiles);
  gcn_zero_f32<<<nblk((long long)N * 64, B), B, 0, stream>>>(agg1, (long long)N * 64);
  gcn_agg1 <<<nblk(E * 16, B), B, 0, stream>>>(src, dst, enorm, H1, agg1, E);
  gcn_post1<<<nblk((long long)N * 64, B), B, 0, stream>>>(agg1, H1, selfn, b1, Hb, N);

  // layer 2
  gcn_gemm2<<<gemmGrid, B, 0, stream>>>(Hb, W2t, H2, N, ntiles);
  gcn_zero_f32<<<nblk((long long)N * 2, B), B, 0, stream>>>(agg2, (long long)N * 2);
  gcn_agg2 <<<nblk(E, B), B, 0, stream>>>(src, dst, enorm, H2, agg2, E);
  gcn_post2<<<nblk((long long)N * 2, B), B, 0, stream>>>(agg2, H2, selfn, b2, out, N);
}
